// H_Encoder_36515811950953
// MI455X (gfx1250) — matrix operand order repaired and checked by reading, not compiled, not run on hardware
//
#include <hip/hip_runtime.h>
#include <hip/hip_bf16.h>

typedef _Float16 h16;
typedef __attribute__((ext_vector_type(16))) _Float16 v16h;
typedef __attribute__((ext_vector_type(8)))  _Float16 v8h;
typedef __attribute__((ext_vector_type(8)))  float    v8f;

#define NB 16          // batch
#define NHEADS 6

union AFrag { v16h v; h16 e[16]; };
union CFrag { v8f  v; float e[8]; };

// ---------------- WMMA fragment loaders (contiguous k = 16*(lane/16) + s, same for A and B) ----------------
// guarded A loader (edge tiles / tiny K)
__device__ __forceinline__ v16h load_a_frag(const h16* A, int lda, int m0, int k0, int Kd, int Md) {
  int lane = threadIdx.x & 31;
  int r = m0 + (lane & 15);
  int kb = k0 + ((lane & 16) ? 16 : 0);
  bool rok = (r < Md);
  const h16* row = A + (size_t)r * lda;
  AFrag f;
#pragma unroll
  for (int i = 0; i < 8; ++i) {
    int k = kb + 2 * i;
    f.e[2 * i]     = (rok && k     < Kd) ? row[k]     : (h16)0.f;
    f.e[2 * i + 1] = (rok && k + 1 < Kd) ? row[k + 1] : (h16)0.f;
  }
  return f.v;
}
// fast A: fully in-bounds tile, lda%8==0 -> two 16B loads per lane
__device__ __forceinline__ v16h load_a_frag_fast(const h16* A, int lda, int m0, int k0) {
  int lane = threadIdx.x & 31;
  const h16* row = A + (size_t)(m0 + (lane & 15)) * lda + k0 + ((lane & 16) ? 16 : 0);
  AFrag f;
  *(v8h*)&f.e[0] = *(const v8h*)row;        // K = kb .. kb+7
  *(v8h*)&f.e[8] = *(const v8h*)(row + 8);  // K = kb+8 .. kb+15
  return f.v;
}
__device__ __forceinline__ v16h load_a_any(const h16* A, int lda, int m0, int k0, int Kd, int Md) {
  if (((lda & 7) == 0) && (m0 + 16 <= Md) && (k0 + 32 <= Kd))
    return load_a_frag_fast(A, lda, m0, k0);
  return load_a_frag(A, lda, m0, k0, Kd, Md);
}
// guarded B loader from W stored (N,K) row-major (computes A @ W^T)
__device__ __forceinline__ v16h load_b_t(const h16* W, int ldw, int n0, int k0, int Kd, int Nd) {
  int lane = threadIdx.x & 31;
  int n = n0 + (lane & 15);
  int kb = k0 + ((lane & 16) ? 16 : 0);
  bool nok = (n < Nd);
  const h16* row = W + (size_t)n * ldw;
  AFrag f;
#pragma unroll
  for (int i = 0; i < 8; ++i) {
    int k = kb + 2 * i;
    f.e[2 * i]     = (nok && k     < Kd) ? row[k]     : (h16)0.f;
    f.e[2 * i + 1] = (nok && k + 1 < Kd) ? row[k + 1] : (h16)0.f;
  }
  return f.v;
}
// fast B^T: 16 contiguous halves per lane -> two 16B loads
__device__ __forceinline__ v16h load_b_t_fast(const h16* W, int ldw, int n0, int k0) {
  int lane = threadIdx.x & 31;
  const h16* row = W + (size_t)(n0 + (lane & 15)) * ldw + k0 + ((lane & 16) ? 16 : 0);
  AFrag f;
  *(v8h*)&f.e[0] = *(const v8h*)row;
  *(v8h*)&f.e[8] = *(const v8h*)(row + 8);
  return f.v;
}
__device__ __forceinline__ v16h load_bt_any(const h16* W, int ldw, int n0, int k0, int Kd, int Nd) {
  if (((ldw & 7) == 0) && (n0 + 16 <= Nd) && (k0 + 32 <= Kd))
    return load_b_t_fast(W, ldw, n0, k0);
  return load_b_t(W, ldw, n0, k0, Kd, Nd);
}
// B from Bm stored (K,N) row-major (natural, computes A @ B); strided -> scalar
__device__ __forceinline__ v16h load_b_n(const h16* Bm, int ldb, int n0, int k0, int Kd, int Nd) {
  int lane = threadIdx.x & 31;
  int n = n0 + (lane & 15);
  int kb = k0 + ((lane & 16) ? 16 : 0);
  bool nok = (n < Nd);
  AFrag f;
#pragma unroll
  for (int i = 0; i < 8; ++i) {
    int k = kb + 2 * i;
    f.e[2 * i]     = (nok && k     < Kd) ? Bm[(size_t)k * ldb + n]       : (h16)0.f;
    f.e[2 * i + 1] = (nok && k + 1 < Kd) ? Bm[(size_t)(k + 1) * ldb + n] : (h16)0.f;
  }
  return f.v;
}

// guarded epilogue: (+bias) -> (*bn_g'+bn_b) -> act -> (+residual) -> store f16/f32
__device__ __forceinline__ void epilogue_store(CFrag acc, h16* C16, float* C32, int ldc,
    int m0, int n0, int M, int N,
    const float* bias, const float* bng, const float* bnb,
    const h16* resid, int ldr, int act) {
  const float RNORM = 0.9999950000f;  // 1/sqrt(1+1e-5)
  int lane = threadIdx.x & 31;
  int col = n0 + (lane & 15);
  int rbase = m0 + ((lane & 16) ? 8 : 0);
  if (col >= N) return;
  float bs = bias ? bias[col] : 0.f;
  float gg = bng ? bng[col] * RNORM : 1.f;
  float bb = bnb ? bnb[col] : 0.f;
#pragma unroll
  for (int r = 0; r < 8; ++r) {
    int row = rbase + r;
    if (row >= M) continue;
    float f = acc.e[r] + bs;
    if (bng) f = f * gg + bb;
    if (act == 1) f = fmaxf(f, 0.f);
    else if (act == 2) f = 0.5f * f * (1.f + erff(f * 0.70710678118f));
    if (resid) f += (float)resid[(size_t)row * ldr + col];
    if (C16) C16[(size_t)row * ldc + col] = (h16)f;
    if (C32) C32[(size_t)row * ldc + col] = f;
  }
}
// unguarded epilogue for the no-edge fast GEMM
__device__ __forceinline__ void epilogue_fast(CFrag acc, h16* C, int ldc, int m0, int col,
    const float* bias, const h16* resid, int ldr, int act) {
  int lane = threadIdx.x & 31;
  int rbase = m0 + ((lane & 16) ? 8 : 0);
  float bs = bias ? bias[col] : 0.f;
#pragma unroll
  for (int r = 0; r < 8; ++r) {
    int row = rbase + r;
    float f = acc.e[r] + bs;
    if (act == 1) f = fmaxf(f, 0.f);
    else if (act == 2) f = 0.5f * f * (1.f + erff(f * 0.70710678118f));
    if (resid) f += (float)resid[(size_t)row * ldr + col];
    C[(size_t)row * ldc + col] = (h16)f;
  }
}

// ---------------- fast GEMM: M%16==0, N%32==0, K%32==0, lda/ldw%8==0, B=W(N,K) ----------------
__global__ void gemm_fast(const h16* __restrict__ A, int lda,
                          const h16* __restrict__ W, int ldw,
                          const float* bias, const h16* resid, int ldr,
                          h16* __restrict__ C, int ldc,
                          int M, int N, int K, int act) {
  int wave = threadIdx.x >> 5, nw = blockDim.x >> 5;
  int tile = blockIdx.x * nw + wave;
  int nt2 = N >> 5;
  if (tile >= (M >> 4) * nt2) return;
  int m0 = (tile / nt2) << 4, n0 = (tile % nt2) << 5;
  CFrag acc0, acc1;
#pragma unroll
  for (int i = 0; i < 8; ++i) { acc0.e[i] = 0.f; acc1.e[i] = 0.f; }
  for (int k0 = 0; k0 < K; k0 += 32) {
    if (k0 + 32 < K) __builtin_prefetch(W + (size_t)n0 * ldw + k0 + 32, 0, 1);
    v16h a  = load_a_frag_fast(A, lda, m0, k0);
    v16h b0 = load_b_t_fast(W, ldw, n0, k0);
    v16h b1 = load_b_t_fast(W, ldw, n0 + 16, k0);
    acc0.v = __builtin_amdgcn_wmma_f32_16x16x32_f16(false, a, false, b0, (short)0, acc0.v, false, false);
    acc1.v = __builtin_amdgcn_wmma_f32_16x16x32_f16(false, a, false, b1, (short)0, acc1.v, false, false);
  }
  int col = n0 + (threadIdx.x & 15);
  epilogue_fast(acc0, C, ldc, m0, col,      bias, resid, ldr, act);
  epilogue_fast(acc1, C, ldc, m0, col + 16, bias, resid, ldr, act);
}

// ---------------- generic guarded GEMM (only pos-MLP layer 1, K=3) ----------------
__global__ void gemm_g(const h16* __restrict__ A, int lda,
                       const h16* __restrict__ Bm, int ldb,
                       const float* bias, h16* C16, int ldc,
                       int M, int N, int K, int act) {
  int wave = threadIdx.x >> 5, nw = blockDim.x >> 5;
  int tile = blockIdx.x * nw + wave;
  int mt = (M + 15) >> 4, nt = (N + 15) >> 4;
  if (tile >= mt * nt) return;
  int m0 = (tile / nt) << 4, n0 = (tile % nt) << 4;
  CFrag acc;
#pragma unroll
  for (int i = 0; i < 8; ++i) acc.e[i] = 0.f;
  for (int k0 = 0; k0 < K; k0 += 32) {
    v16h a = load_a_frag(A, lda, m0, k0, K, M);
    v16h b = load_b_t(Bm, ldb, n0, k0, K, N);
    acc.v = __builtin_amdgcn_wmma_f32_16x16x32_f16(false, a, false, b, (short)0, acc.v, false, false);
  }
  epilogue_store(acc, C16, nullptr, ldc, m0, n0, M, N, bias, nullptr, nullptr, nullptr, 0, act);
}

// ---------------- batched attention GEMMs, templated on model dim ----------------
template <int D>
__global__ void attn_scores_k(const h16* __restrict__ qkv, float* __restrict__ S, int L) {
  const int dh = D / NHEADS;
  int bh = blockIdx.y, b = bh / NHEADS, h = bh % NHEADS;
  const h16* Q  = qkv + (size_t)b * L * 3 * D + (size_t)h * dh;
  const h16* Kp = qkv + (size_t)b * L * 3 * D + D + (size_t)h * dh;
  float* Sp = S + (size_t)bh * L * L;
  int wave = threadIdx.x >> 5, nw = blockDim.x >> 5;
  int tile = blockIdx.x * nw + wave;
  int t16 = (L + 15) >> 4;
  if (tile >= t16 * t16) return;
  int m0 = (tile / t16) << 4, n0 = (tile % t16) << 4;
  CFrag acc;
#pragma unroll
  for (int i = 0; i < 8; ++i) acc.e[i] = 0.f;
#pragma unroll
  for (int kk = 0; kk < (dh + 31) / 32; ++kk) {
    int k0 = kk * 32;
    v16h a  = load_a_any(Q, 3 * D, m0, k0, dh, L);
    v16h bb = load_bt_any(Kp, 3 * D, n0, k0, dh, L);
    acc.v = __builtin_amdgcn_wmma_f32_16x16x32_f16(false, a, false, bb, (short)0, acc.v, false, false);
  }
  epilogue_store(acc, nullptr, Sp, L, m0, n0, L, L, nullptr, nullptr, nullptr, nullptr, 0, 0);
}

template <int D>
__global__ void attn_out_k(const h16* __restrict__ P, const h16* __restrict__ qkv,
                           h16* __restrict__ O, int L) {
  const int dh = D / NHEADS;
  int bh = blockIdx.y, b = bh / NHEADS, h = bh % NHEADS;
  const h16* A = P + (size_t)bh * L * L;
  const h16* V = qkv + (size_t)b * L * 3 * D + 2 * D + (size_t)h * dh;
  h16* Op = O + (size_t)b * L * D + (size_t)h * dh;
  int wave = threadIdx.x >> 5, nw = blockDim.x >> 5;
  int tile = blockIdx.x * nw + wave;
  const int nt = dh / 16;
  int mt = (L + 15) >> 4;
  if (tile >= mt * nt) return;
  int m0 = (tile / nt) << 4, n0 = (tile % nt) << 4;
  CFrag acc;
#pragma unroll
  for (int i = 0; i < 8; ++i) acc.e[i] = 0.f;
  for (int k0 = 0; k0 < L; k0 += 32) {
    v16h a  = load_a_any(A, L, m0, k0, L, L);
    v16h bb = load_b_n(V, 3 * D, n0, k0, L, dh);
    acc.v = __builtin_amdgcn_wmma_f32_16x16x32_f16(false, a, false, bb, (short)0, acc.v, false, false);
  }
  epilogue_store(acc, Op, nullptr, D, m0, n0, L, dh, nullptr, nullptr, nullptr, nullptr, 0, 0);
}

// ---------------- wave-per-row softmax (unmasked) and LayerNorm ----------------
__global__ void softmax_k(const float* __restrict__ S, h16* __restrict__ P, int L, float scale) {
  size_t row = blockIdx.x;
  const float* s = S + row * L;
  h16* p = P + row * L;
  int lane = threadIdx.x;
  float m = -1e30f;
  for (int j = lane; j < L; j += 32) m = fmaxf(m, s[j] * scale);
  for (int o = 16; o > 0; o >>= 1) m = fmaxf(m, __shfl_xor(m, o, 32));
  float sum = 0.f;
  for (int j = lane; j < L; j += 32) sum += expf(s[j] * scale - m);
  for (int o = 16; o > 0; o >>= 1) sum += __shfl_xor(sum, o, 32);
  float inv = 1.f / sum;
  for (int j = lane; j < L; j += 32) p[j] = (h16)(expf(s[j] * scale - m) * inv);
}

__global__ void ln_k(const h16* __restrict__ X, const float* __restrict__ g,
                     const float* __restrict__ b, h16* Y16, float* Y32, int d) {
  size_t row = blockIdx.x;
  const h16* x = X + row * d;
  int lane = threadIdx.x;
  float s = 0.f, s2 = 0.f;
  for (int j = lane; j < d; j += 32) { float v = (float)x[j]; s += v; s2 += v * v; }
  for (int o = 16; o > 0; o >>= 1) { s += __shfl_xor(s, o, 32); s2 += __shfl_xor(s2, o, 32); }
  float mean = s / d;
  float var = s2 / d - mean * mean;
  float inv = rsqrtf(var + 1e-5f);
  for (int j = lane; j < d; j += 32) {
    float v = ((float)x[j] - mean) * inv * g[j] + b[j];
    if (Y16) Y16[row * d + j] = (h16)v;
    if (Y32) Y32[row * d + j] = v;
  }
}

// ---------------- fused per-group PointNet token embed (all in LDS) ----------------
// template dims -> fast/guarded path chosen at compile time; lda==ldw==K
template <int M, int N, int K>
__device__ __forceinline__ void lds_gemm_t(const h16* Ash, const h16* Wg,
    const float* bias, const float* bng, const float* bnb, int act,
    h16* Csh, int nw, int wave) {
  const int mt = M / 16, nt = N / 16, kt = (K + 31) / 32;
  for (int t = wave; t < mt * nt; t += nw) {
    int m0 = (t / nt) << 4, n0 = (t % nt) << 4;
    CFrag acc;
#pragma unroll
    for (int i = 0; i < 8; ++i) acc.e[i] = 0.f;
#pragma unroll
    for (int kk = 0; kk < kt; ++kk) {
      const int k0 = kk * 32;
      v16h a = ((K % 8) == 0 && k0 + 32 <= K) ? load_a_frag_fast(Ash, K, m0, k0)
                                              : load_a_frag(Ash, K, m0, k0, K, M);
      v16h b = ((K % 8) == 0 && k0 + 32 <= K) ? load_b_t_fast(Wg, K, n0, k0)
                                              : load_b_t(Wg, K, n0, k0, K, N);
      acc.v = __builtin_amdgcn_wmma_f32_16x16x32_f16(false, a, false, b, (short)0, acc.v, false, false);
    }
    epilogue_store(acc, Csh, nullptr, N, m0, n0, M, N, bias, bng, bnb, nullptr, 0, act);
  }
}

template <int KNT, int INCT, int HDT, int DDT>
__global__ void token_embed_k(const float* __restrict__ pts32,
                              const h16* __restrict__ tokPrev, const int* __restrict__ idx,
                              const h16* w1, const float* b1, const float* bn1g, const float* bn1b,
                              const h16* w2, const float* b2,
                              const h16* w3, const float* b3, const float* bn2g, const float* bn2b,
                              const h16* w4, const float* b4,
                              h16* __restrict__ tokOut) {
  __shared__ __align__(16) h16 sX[KNT * INCT];
  __shared__ __align__(16) h16 sH1[KNT * HDT];
  __shared__ __align__(16) h16 sH2[KNT * HDT];
  __shared__ __align__(16) h16 sCC[KNT * DDT];
  __shared__ __align__(16) h16 sH3[KNT * DDT];
  __shared__ __align__(16) h16 sH4[KNT * DDT];
  int grp = blockIdx.x, tid = threadIdx.x;
  if (pts32) {
    const float* src = pts32 + (size_t)grp * KNT * INCT;
    for (int i = tid; i < KNT * INCT; i += blockDim.x) sX[i] = (h16)src[i];
  } else {
    for (int i = tid; i < KNT * INCT; i += blockDim.x) {
      int k = i / INCT, c = i % INCT;
      int row = idx[(size_t)grp * KNT + k];
      sX[i] = tokPrev[(size_t)row * INCT + c];
    }
  }
  __syncthreads();
  int wave = tid >> 5, nw = blockDim.x >> 5;
  lds_gemm_t<KNT, HDT, INCT>(sX, w1, b1, bn1g, bn1b, 1, sH1, nw, wave);
  __syncthreads();
  lds_gemm_t<KNT, HDT, HDT>(sH1, w2, b2, nullptr, nullptr, 0, sH2, nw, wave);
  __syncthreads();
  for (int c = tid; c < HDT; c += blockDim.x) {     // max-pool + concat [gl, h]
    float m = -1e30f;
    for (int k = 0; k < KNT; ++k) m = fmaxf(m, (float)sH2[k * HDT + c]);
    for (int k = 0; k < KNT; ++k) { sCC[k * DDT + c] = (h16)m; sCC[k * DDT + HDT + c] = sH2[k * HDT + c]; }
  }
  __syncthreads();
  lds_gemm_t<KNT, DDT, DDT>(sCC, w3, b3, bn2g, bn2b, 1, sH3, nw, wave);
  __syncthreads();
  lds_gemm_t<KNT, DDT, DDT>(sH3, w4, b4, nullptr, nullptr, 0, sH4, nw, wave);
  __syncthreads();
  for (int c = tid; c < DDT; c += blockDim.x) {
    float m = -1e30f;
    for (int k = 0; k < KNT; ++k) m = fmaxf(m, (float)sH4[k * DDT + c]);
    tokOut[(size_t)grp * DDT + c] = (h16)m;
  }
}

// ---------------- small utility kernels ----------------
__global__ void cvt_k(const float* __restrict__ s, h16* __restrict__ d, int n) {
  int i = blockIdx.x * blockDim.x + threadIdx.x;
  if (i < n) d[i] = (h16)s[i];
}
__global__ void fill_k(float* p, float v, int n) {
  int i = blockIdx.x * blockDim.x + threadIdx.x;
  if (i < n) p[i] = v;
}
__global__ void add_k(const v8h* a, const v8h* b, v8h* o, int n8) {
  int i = blockIdx.x * blockDim.x + threadIdx.x;
  if (i >= n8) return;
  v8h x = a[i], y = b[i];
  v8h z;
#pragma unroll
  for (int j = 0; j < 8; ++j) z[j] = (h16)((float)x[j] + (float)y[j]);
  o[i] = z;
}
// stable rank of -sals; mask = rank >= 26 (level 2)
__global__ void rank_mask_k(const float* __restrict__ sals, float* __restrict__ mask2) {
  int b = blockIdx.x, g = threadIdx.x;               // 128 threads
  float v = sals[b * 128 + g];
  int rank = 0;
  for (int j = 0; j < 128; ++j) {
    float u = sals[b * 128 + j];
    rank += (u > v) || (u == v && j < g);
  }
  mask2[b * 128 + g] = (rank >= 26) ? 1.f : 0.f;
}
// replicate reference scatter: idx_masked = keep ? idx : 0; parent[idx_masked] = 0
__global__ void scatter_mask_k(const float* child, const int* idxs, float* parent,
                               int nChild, int knb) {
  int e = blockIdx.x * blockDim.x + threadIdx.x;
  if (e >= nChild * knb) return;
  int ce = e / knb;
  bool keep = child[ce] < 0.5f;
  int t = keep ? idxs[e] : 0;
  parent[t] = 0.f;
}
__global__ void compact_k(const float* mask, int G, int L, int* ordv) {
  int b = blockIdx.x * blockDim.x + threadIdx.x;
  if (b >= NB) return;
  int cnt = 0;
  for (int g = 0; g < G && cnt < L; ++g)
    if (mask[b * G + g] < 0.5f) ordv[b * L + cnt++] = g;
  while (cnt < L) ordv[b * L + cnt++] = 0;
}
// gather/scatter rows in 8-half (16B) packets; d is a multiple of 8
__global__ void gather_rows_k(const h16* T, const int* ordv, h16* X, int G, int L, int d) {
  size_t i = (size_t)blockIdx.x * blockDim.x + threadIdx.x;   // packet index
  int d8 = d >> 3;
  size_t total = (size_t)NB * L * d8;
  if (i >= total) return;
  int c = i % d8; size_t r = i / d8; int l = r % L; int b = r / L;
  const v8h* src = (const v8h*)(T + ((size_t)b * G + ordv[b * L + l]) * d);
  ((v8h*)X)[i] = src[c];
}
__global__ void scatter_rows_k(h16* T, const int* ordv, const h16* X, int G, int L, int d) {
  size_t i = (size_t)blockIdx.x * blockDim.x + threadIdx.x;
  int d8 = d >> 3;
  size_t total = (size_t)NB * L * d8;
  if (i >= total) return;
  int c = i % d8; size_t r = i / d8; int l = r % L; int b = r / L;
  v8h* dst = (v8h*)(T + ((size_t)b * G + ordv[b * L + l]) * d);
  dst[c] = ((const v8h*)X)[i];
}
__global__ void gather_centers_k(const float* C, const int* ordv, h16* out, int G, int L) {
  size_t i = (size_t)blockIdx.x * blockDim.x + threadIdx.x;
  size_t total = (size_t)NB * L * 3;
  if (i >= total) return;
  int c = i % 3; size_t r = i / 3; int l = r % L; int b = r / L;
  out[i] = (h16)C[((size_t)b * G + ordv[b * L + l]) * 3 + c];
}

// ================================================================================
extern "C" void kernel_launch(void* const* d_in, const int* in_sizes, int n_in,
                              void* d_out, int out_size, void* d_ws, size_t ws_size,
                              hipStream_t stream) {
  (void)in_sizes; (void)n_in; (void)out_size; (void)ws_size;
  const int DIMS[3] = {96, 192, 384}, INC[3] = {3, 96, 192}, HD[3] = {48, 96, 192};
  const int Gs[3] = {2048, 512, 128}, Ls[3] = {416, 104, 26};

  const float* neigh0 = (const float*)d_in[0];
  const float* centers[3] = {(const float*)d_in[3], (const float*)d_in[4], (const float*)d_in[5]};
  const int* idxs1 = (const int*)d_in[6];
  const int* idxs2 = (const int*)d_in[7];
  const float* sals = (const float*)d_in[8];
  // params pytree flattened with sorted dict keys: blocks, norms, pos, te
  auto P  = [&](int i) { return (const float*)d_in[i]; };
  auto PB = [&](int l, int t, int o) { return P(9 + (l * 5 + t) * 11 + o); };   // fc1_b,fc1_w,fc2_b,fc2_w,ln1_b,ln1_g,ln2_b,ln2_g,proj_b,proj_w,qkv_w
  auto PN = [&](int j, int o) { return P(174 + j * 2 + o); };                    // b,g
  auto PP = [&](int j, int o) { return P(180 + j * 4 + o); };                    // b1,b2,w1,w2
  auto PT = [&](int j, int o) { return P(192 + j * 12 + o); };                   // b1..b4,bn1_b,bn1_g,bn2_b,bn2_g,w1..w4

  char* base = (char*)d_ws; size_t off = 0;
  auto alloc  = [&](size_t bytes) { void* p = base + off; off = (off + bytes + 255) & ~(size_t)255; return p; };
  auto allocH = [&](size_t n) { return (h16*)alloc(n * sizeof(h16)); };
  auto allocF = [&](size_t n) { return (float*)alloc(n * sizeof(float)); };
  auto allocI = [&](size_t n) { return (int*)alloc(n * sizeof(int)); };
  auto cvt = [&](const float* s, size_t n) {
    h16* d = allocH(n);
    cvt_k<<<(int)((n + 255) / 256), 256, 0, stream>>>(s, d, (int)n);
    return d;
  };

  // convert all weights to f16 in ws
  h16 *teW1[3], *teW2[3], *teW3[3], *teW4[3], *posW1[3], *posW2[3];
  h16 *qkvW[3][5], *projW[3][5], *fc1W[3][5], *fc2W[3][5];
  for (int j = 0; j < 3; ++j) {
    teW1[j]  = cvt(PT(j, 8),  (size_t)HD[j] * INC[j]);
    teW2[j]  = cvt(PT(j, 9),  (size_t)HD[j] * HD[j]);
    teW3[j]  = cvt(PT(j, 10), (size_t)DIMS[j] * DIMS[j]);
    teW4[j]  = cvt(PT(j, 11), (size_t)DIMS[j] * DIMS[j]);
    posW1[j] = cvt(PP(j, 2),  (size_t)DIMS[j] * 3);
    posW2[j] = cvt(PP(j, 3),  (size_t)DIMS[j] * DIMS[j]);
    for (int t = 0; t < 5; ++t) {
      int d = DIMS[j];
      qkvW[j][t]  = cvt(PB(j, t, 10), (size_t)3 * d * d);
      projW[j][t] = cvt(PB(j, t, 9),  (size_t)d * d);
      fc1W[j][t]  = cvt(PB(j, t, 1),  (size_t)4 * d * d);
      fc2W[j][t]  = cvt(PB(j, t, 3),  (size_t)4 * d * d);
    }
  }

  // activation workspace (sized for level 0, the largest)
  h16* tokens[3];
  for (int j = 0; j < 3; ++j) tokens[j] = allocH((size_t)NB * Gs[j] * DIMS[j]);
  const size_t XE = (size_t)NB * 416 * 96;
  h16* xbuf = allocH(XE);  h16* xp = allocH(XE);  h16* xln = allocH(XE);
  h16* qkvb = allocH(3 * XE);
  h16* ob = allocH(XE);
  h16* hid = allocH((size_t)NB * 416 * 384);
  float* scores = allocF((size_t)NB * NHEADS * 416 * 416);
  h16* probs = allocH((size_t)NB * NHEADS * 416 * 416);
  h16* posb = allocH(XE); h16* poshid = allocH(XE); h16* posc = allocH((size_t)NB * 416 * 3);
  int* ordv[3];
  for (int j = 0; j < 3; ++j) ordv[j] = allocI((size_t)NB * 416);

  // d_out layout: outs0,outs1,outs2, mvis0..2, masks0..2 (all as f32)
  float* out32 = (float*)d_out;
  float* outs[3];
  outs[0] = out32;
  outs[1] = outs[0] + (size_t)NB * 416 * 96;
  outs[2] = outs[1] + (size_t)NB * 104 * 192;
  float* mvisO[3];
  mvisO[0] = outs[2] + (size_t)NB * 26 * 384;
  mvisO[1] = mvisO[0] + NB * 416;
  mvisO[2] = mvisO[1] + NB * 104;
  float* masksO[3];
  masksO[0] = mvisO[2] + NB * 26;
  masksO[1] = masksO[0] + NB * 2048;
  masksO[2] = masksO[1] + NB * 512;

  // fast GEMM launcher: M%16==0, N%32==0, K%32==0
  auto gemmF = [&](const h16* A, int lda, const h16* W, int ldw,
                   const float* bias, const h16* res, int ldr,
                   h16* C, int ldc, int M, int N, int K, int act) {
    int tiles = (M / 16) * (N / 32);
    gemm_fast<<<(tiles + 7) / 8, 256, 0, stream>>>(A, lda, W, ldw, bias, res, ldr,
                                                   C, ldc, M, N, K, act);
  };

  // ---- masks ----
  rank_mask_k<<<NB, 128, 0, stream>>>(sals, masksO[2]);
  fill_k<<<(NB * 512 + 255) / 256, 256, 0, stream>>>(masksO[1], 1.f, NB * 512);
  scatter_mask_k<<<(NB * 128 * 16 + 255) / 256, 256, 0, stream>>>(masksO[2], idxs2, masksO[1], NB * 128, 16);
  fill_k<<<(NB * 2048 + 255) / 256, 256, 0, stream>>>(masksO[0], 1.f, NB * 2048);
  scatter_mask_k<<<(NB * 512 * 16 + 255) / 256, 256, 0, stream>>>(masksO[1], idxs1, masksO[0], NB * 512, 16);
  for (int j = 0; j < 3; ++j) {
    compact_k<<<1, NB, 0, stream>>>(masksO[j], Gs[j], Ls[j], ordv[j]);
    fill_k<<<(NB * Ls[j] + 255) / 256, 256, 0, stream>>>(mvisO[j], 1.f, NB * Ls[j]);
  }

  // ---- per-level pipeline ----
  for (int i = 0; i < 3; ++i) {
    int d = DIMS[i], L = Ls[i], dh = d / NHEADS, G = Gs[i];
    int rows = NB * L;
    size_t nel = (size_t)rows * d;
    size_t npkt = nel >> 3;                 // 8-half packets
    // token embedding (fused PointNet in LDS)
    if (i == 0)
      token_embed_k<32, 3, 48, 96><<<NB * 2048, 128, 0, stream>>>(
          neigh0, nullptr, nullptr,
          teW1[0], PT(0, 0), PT(0, 5), PT(0, 4), teW2[0], PT(0, 1),
          teW3[0], PT(0, 2), PT(0, 7), PT(0, 6), teW4[0], PT(0, 3), tokens[0]);
    else if (i == 1)
      token_embed_k<16, 96, 96, 192><<<NB * 512, 128, 0, stream>>>(
          nullptr, tokens[0], idxs1,
          teW1[1], PT(1, 0), PT(1, 5), PT(1, 4), teW2[1], PT(1, 1),
          teW3[1], PT(1, 2), PT(1, 7), PT(1, 6), teW4[1], PT(1, 3), tokens[1]);
    else
      token_embed_k<16, 192, 192, 384><<<NB * 128, 128, 0, stream>>>(
          nullptr, tokens[1], idxs2,
          teW1[2], PT(2, 0), PT(2, 5), PT(2, 4), teW2[2], PT(2, 1),
          teW3[2], PT(2, 2), PT(2, 7), PT(2, 6), teW4[2], PT(2, 3), tokens[2]);

    gather_rows_k<<<(int)((npkt + 255) / 256), 256, 0, stream>>>(tokens[i], ordv[i], xbuf, G, L, d);

    // positional MLP: gelu(c@w1^T+b1)@w2^T+b2  (layer1 K=3 -> guarded gemm)
    gather_centers_k<<<(NB * L * 3 + 255) / 256, 256, 0, stream>>>(centers[i], ordv[i], posc, G, L);
    {
      int tiles = ((rows + 15) / 16) * ((d + 15) / 16);
      gemm_g<<<(tiles + 7) / 8, 256, 0, stream>>>(posc, 3, posW1[i], 3, PP(i, 0), poshid, d, rows, d, 3, 2);
    }
    gemmF(poshid, d, posW2[i], d, PP(i, 1), nullptr, 0, posb, d, rows, d, d, 0);

    int t16 = (L + 15) / 16;
    int stiles = t16 * t16;
    int otiles = t16 * (dh / 16);
    for (int t = 0; t < 5; ++t) {
      add_k<<<(int)((npkt + 255) / 256), 256, 0, stream>>>((const v8h*)xbuf, (const v8h*)posb, (v8h*)xp, (int)npkt);
      ln_k<<<rows, 32, 0, stream>>>(xp, PB(i, t, 5), PB(i, t, 4), xln, nullptr, d);
      gemmF(xln, d, qkvW[i][t], d, nullptr, nullptr, 0, qkvb, 3 * d, rows, 3 * d, d, 0);
      if (i == 0)      attn_scores_k<96><<<dim3((stiles + 7) / 8, NB * NHEADS), 256, 0, stream>>>(qkvb, scores, L);
      else if (i == 1) attn_scores_k<192><<<dim3((stiles + 7) / 8, NB * NHEADS), 256, 0, stream>>>(qkvb, scores, L);
      else             attn_scores_k<384><<<dim3((stiles + 7) / 8, NB * NHEADS), 256, 0, stream>>>(qkvb, scores, L);
      softmax_k<<<NB * NHEADS * L, 32, 0, stream>>>(scores, probs, L, rsqrtf((float)dh));
      if (i == 0)      attn_out_k<96><<<dim3((otiles + 7) / 8, NB * NHEADS), 256, 0, stream>>>(probs, qkvb, ob, L);
      else if (i == 1) attn_out_k<192><<<dim3((otiles + 7) / 8, NB * NHEADS), 256, 0, stream>>>(probs, qkvb, ob, L);
      else             attn_out_k<384><<<dim3((otiles + 7) / 8, NB * NHEADS), 256, 0, stream>>>(probs, qkvb, ob, L);
      gemmF(ob, d, projW[i][t], d, PB(i, t, 8), xp, d, xbuf, d, rows, d, d, 0);
      ln_k<<<rows, 32, 0, stream>>>(xbuf, PB(i, t, 7), PB(i, t, 6), xln, nullptr, d);
      gemmF(xln, d, fc1W[i][t], d, PB(i, t, 0), nullptr, 0, hid, 4 * d, rows, 4 * d, d, 2);
      gemmF(hid, 4 * d, fc2W[i][t], 4 * d, PB(i, t, 2), xbuf, d, xbuf, d, rows, d, 4 * d, 0);
    }
    ln_k<<<rows, 32, 0, stream>>>(xbuf, PN(i, 1), PN(i, 0), nullptr, outs[i], d);
    if (i < 2)
      scatter_rows_k<<<(int)((npkt + 255) / 256), 256, 0, stream>>>(tokens[i], ordv[i], xbuf, G, L, d);
  }
}
